// GatedAttention_84293028152097
// MI455X (gfx1250) — matrix operand order repaired and checked by reading, not compiled, not run on hardware
//
#include <hip/hip_runtime.h>
#include <hip/hip_bf16.h>
#include <math.h>

#define HIDDEN 768
#define KEY    128
#define INTER  1536
#define BATCH  4
#define SEQ    2048
#define NTOK   (BATCH * SEQ)      /* 8192 */
#define NI     (2 * INTER + KEY)  /* 3200 */

typedef __attribute__((ext_vector_type(16))) __bf16        v16bf;
typedef __attribute__((ext_vector_type(8)))  float         v8f;
typedef __attribute__((ext_vector_type(4)))  unsigned int  uint4v;
typedef unsigned short BF16;

// float -> bf16 with round-to-nearest-even
__device__ __forceinline__ BF16 f2bf(float f) {
  union { float f; unsigned int u; } v; v.f = f;
  unsigned int r = v.u + 0x7FFFu + ((v.u >> 16) & 1u);
  return (BF16)(r >> 16);
}

union Frag16 { uint4v q[2]; v16bf v; };

// A fragment (16x32 bf16, row-major source, lda in elements).
// ISA layout: lanes 0-15 hold M=lane, K = {8h..8h+7, 16+8h..16+8h+7}
__device__ __forceinline__ v16bf load_frag_a(const BF16* A, int lda, long m0, int k0, int lane) {
  int h = lane >> 4, m = lane & 15;
  const BF16* p = A + (m0 + m) * (long)lda + k0 + 8 * h;
  Frag16 f;
  f.q[0] = *reinterpret_cast<const uint4v*>(p);
  f.q[1] = *reinterpret_cast<const uint4v*>(p + 16);
  return f.v;
}

// B fragment (32x16 bf16). Source laid out so fixed-N rows are contiguous in K
// (i.e. BT[n][k]).  ISA layout: lane (h,n) holds K = {8h..8h+7, 16+8h..16+8h+7}.
__device__ __forceinline__ v16bf load_frag_b(const BF16* BT, int ldb, long n0, int k0, int lane) {
  int h = lane >> 4, n = lane & 15;
  const BF16* p = BT + (n0 + n) * (long)ldb + k0 + 8 * h;
  Frag16 f;
  f.q[0] = *reinterpret_cast<const uint4v*>(p);
  f.q[1] = *reinterpret_cast<const uint4v*>(p + 16);
  return f.v;
}

#define WMMA_BF16(a, b, c) \
  __builtin_amdgcn_wmma_f32_16x16x32_bf16(false, (a), false, (b), (short)0, (c), false, false)

__device__ __forceinline__ float silu(float x) { return x / (1.0f + __expf(-x)); }

// ---------------- pack / prep kernels ----------------

__global__ void k_cvt_bf16(const float* __restrict__ src, BF16* __restrict__ dst, long n) {
  long i = (long)blockIdx.x * blockDim.x + threadIdx.x;
  if (i < n) dst[i] = f2bf(src[i]);
}

__global__ void k_transpose_bf16(const float* __restrict__ src, BF16* __restrict__ dst,
                                 int R, int C) {
  long i = (long)blockIdx.x * blockDim.x + threadIdx.x;
  long total = (long)R * C;
  if (i < total) {
    int r = (int)(i / C), c = (int)(i % C);
    dst[(long)c * R + r] = f2bf(src[i]);
  }
}

// per-batch l = max(sum(mask), 1);  scale = log(l)/log(512)
__global__ void k_scale(const int* __restrict__ mask, float* __restrict__ scaleB) {
  __shared__ int red[256];
  int b = blockIdx.x, t = threadIdx.x;
  int cnt = 0;
  for (int j = t; j < SEQ; j += 256) cnt += (mask[b * SEQ + j] != 0) ? 1 : 0;
  red[t] = cnt;
  __syncthreads();
  for (int s = 128; s > 0; s >>= 1) {
    if (t < s) red[t] += red[t + s];
    __syncthreads();
  }
  if (t == 0) {
    float l = fmaxf((float)red[0], 1.0f);
    scaleB[b] = logf(l) / logf(512.0f);
  }
}

// ---------------- GEMM1: h = silu(X*Wi + bi); split u / vT / q / k ----------------
// One wave computes a 32x64 tile (2 M-tiles x 4 N-tiles, 8 WMMA accumulators).

__global__ __launch_bounds__(32) void k_gemm1(
    const BF16* __restrict__ Xb, const BF16* __restrict__ WiT,
    const float* __restrict__ bi, const float* __restrict__ gamma, const float* __restrict__ beta,
    float* __restrict__ u, BF16* __restrict__ vT,
    BF16* __restrict__ qb, BF16* __restrict__ kb) {
  int lane = threadIdx.x;
  int n0 = blockIdx.x * 64;
  long m0 = (long)blockIdx.y * 32;
  int h = lane >> 4, ln = lane & 15;

  v8f acc[2][4];
#pragma unroll
  for (int i = 0; i < 2; ++i)
#pragma unroll
    for (int j = 0; j < 4; ++j) {
      float bias = bi[n0 + j * 16 + ln];
      acc[i][j] = (v8f){bias, bias, bias, bias, bias, bias, bias, bias};
    }

  for (int k0 = 0; k0 < HIDDEN; k0 += 32) {
    v16bf a0 = load_frag_a(Xb, HIDDEN, m0, k0, lane);
    v16bf a1 = load_frag_a(Xb, HIDDEN, m0 + 16, k0, lane);
    if (k0 + 64 < HIDDEN) {  // distance-2 prefetch -> global_prefetch_b8
      __builtin_prefetch(Xb + (m0 + (lane & 15)) * (long)HIDDEN + k0 + 64, 0, 1);
      __builtin_prefetch(WiT + (n0 + (lane & 15)) * (long)HIDDEN + k0 + 64, 0, 1);
    }
#pragma unroll
    for (int j = 0; j < 4; ++j) {
      v16bf b = load_frag_b(WiT, HIDDEN, n0 + j * 16, k0, lane);
      acc[0][j] = WMMA_BF16(a0, b, acc[0][j]);
      acc[1][j] = WMMA_BF16(a1, b, acc[1][j]);
    }
  }

  if (n0 < INTER) {
#pragma unroll
    for (int i = 0; i < 2; ++i) {
      long rbase = m0 + i * 16 + 8 * h;
#pragma unroll
      for (int j = 0; j < 4; ++j) {
        int col = n0 + j * 16 + ln;
#pragma unroll
        for (int r = 0; r < 8; ++r)
          u[(rbase + r) * (long)INTER + col] = silu(acc[i][j][r]);
      }
    }
  } else if (n0 < 2 * INTER) {
#pragma unroll
    for (int i = 0; i < 2; ++i) {
      long rbase = m0 + i * 16 + 8 * h;
      int bb = (int)(rbase / SEQ);
      int s0 = (int)(rbase % SEQ);
#pragma unroll
      for (int j = 0; j < 4; ++j) {
        int d = n0 + j * 16 + ln - INTER;
        union { BF16 s[8]; uint4v q; } pk;
#pragma unroll
        for (int r = 0; r < 8; ++r) pk.s[r] = f2bf(silu(acc[i][j][r]));
        *reinterpret_cast<uint4v*>(vT + ((long)bb * INTER + d) * SEQ + s0) = pk.q;
      }
    }
  } else {
#pragma unroll
    for (int i = 0; i < 2; ++i) {
      long rbase = m0 + i * 16 + 8 * h;
#pragma unroll
      for (int j = 0; j < 4; ++j) {
        int d = n0 + j * 16 + ln - 2 * INTER;
        float g0 = gamma[d], b0 = beta[d], g1 = gamma[KEY + d], b1 = beta[KEY + d];
#pragma unroll
        for (int r = 0; r < 8; ++r) {
          float sv = silu(acc[i][j][r]);
          long row = rbase + r;
          qb[row * KEY + d] = f2bf(sv * g0 + b0);
          kb[row * KEY + d] = f2bf(sv * g1 + b1);
        }
      }
    }
  }
}

// ---------------- attention: softmax((q k^T/sqrt(K) + am)*scale) @ v, gated by u ----------------

__global__ __launch_bounds__(128) void k_attn(
    const BF16* __restrict__ qb, const BF16* __restrict__ kb, const BF16* __restrict__ vT,
    const int* __restrict__ mask, const float* __restrict__ scaleB,
    const float* __restrict__ u, BF16* __restrict__ uc) {
  __shared__ __align__(16) unsigned char smem[16 * SEQ * 2];  // 64 KB
  float* stats = reinterpret_cast<float*>(smem);              // phase 1: [4][16][2]
  BF16* pbuf = reinterpret_cast<BF16*>(smem);                 // phases 2/3: [16][SEQ]

  const int tid = threadIdx.x;
  const int wave = tid >> 5, lane = tid & 31;
  const int h = lane >> 4, ln = lane & 15;
  const int b = blockIdx.y;
  const long m0 = (long)b * SEQ + (long)blockIdx.x * 16;  // global row of tile
  const float scale = scaleB[b];
  const int* mrowp = mask + b * SEQ;
  const float RSQ = 0.08838834764831845f;  // 1/sqrt(128)

  v16bf aq[4];
#pragma unroll
  for (int kk = 0; kk < 4; ++kk) aq[kk] = load_frag_a(qb, KEY, m0, kk * 32, lane);

  // ---- phase 1: online row max / sum-exp over this wave's j-range ----
  float mr[8], sr[8];
#pragma unroll
  for (int r = 0; r < 8; ++r) { mr[r] = -1e30f; sr[r] = 0.0f; }
  for (int jt = wave * 32; jt < wave * 32 + 32; ++jt) {
    v8f c = {0.f, 0.f, 0.f, 0.f, 0.f, 0.f, 0.f, 0.f};
    long jn0 = (long)b * SEQ + jt * 16;
#pragma unroll
    for (int kk = 0; kk < 4; ++kk)
      c = WMMA_BF16(aq[kk], load_frag_b(kb, KEY, jn0, kk * 32, lane), c);
    int j = jt * 16 + ln;
    float am = (1.0f - (float)mrowp[j]) * -1.0e12f;
#pragma unroll
    for (int r = 0; r < 8; ++r) {
      float sc = (c[r] * RSQ + am) * scale;
      if (sc > mr[r]) { sr[r] = sr[r] * __expf(mr[r] - sc) + 1.0f; mr[r] = sc; }
      else             sr[r] += __expf(sc - mr[r]);
    }
  }
#pragma unroll
  for (int r = 0; r < 8; ++r) {
#pragma unroll
    for (int off = 8; off >= 1; off >>= 1) {  // reduce across 16 lanes of this half
      float om = __shfl_xor(mr[r], off, 16);
      float os = __shfl_xor(sr[r], off, 16);
      float nm = fmaxf(mr[r], om);
      sr[r] = sr[r] * __expf(mr[r] - nm) + os * __expf(om - nm);
      mr[r] = nm;
    }
  }
  if (ln == 0) {
#pragma unroll
    for (int r = 0; r < 8; ++r) {
      int idx = (wave * 16 + r + 8 * h) * 2;
      stats[idx] = mr[r];
      stats[idx + 1] = sr[r];
    }
  }
  __syncthreads();
  float Mf[8], Si[8];
#pragma unroll
  for (int r = 0; r < 8; ++r) {
    int rowl = r + 8 * h;
    float M = -1e30f;
    for (int w = 0; w < 4; ++w) M = fmaxf(M, stats[(w * 16 + rowl) * 2]);
    float S = 0.0f;
    for (int w = 0; w < 4; ++w)
      S += stats[(w * 16 + rowl) * 2 + 1] * __expf(stats[(w * 16 + rowl) * 2] - M);
    Mf[r] = M;
    Si[r] = 1.0f / S;
  }
  __syncthreads();  // done with stats region; reuse as prob tile

  // ---- phase 2: recompute scores, write normalized probs (bf16) to LDS ----
  for (int jt = wave * 32; jt < wave * 32 + 32; ++jt) {
    v8f c = {0.f, 0.f, 0.f, 0.f, 0.f, 0.f, 0.f, 0.f};
    long jn0 = (long)b * SEQ + jt * 16;
#pragma unroll
    for (int kk = 0; kk < 4; ++kk)
      c = WMMA_BF16(aq[kk], load_frag_b(kb, KEY, jn0, kk * 32, lane), c);
    int j = jt * 16 + ln;
    float am = (1.0f - (float)mrowp[j]) * -1.0e12f;
#pragma unroll
    for (int r = 0; r < 8; ++r) {
      float sc = (c[r] * RSQ + am) * scale;
      pbuf[(r + 8 * h) * SEQ + j] = f2bf(__expf(sc - Mf[r]) * Si[r]);
    }
  }
  __syncthreads();

  // ---- phase 3: ctx = P @ V; 4 N-tiles share each LDS A-fragment ----
  const BF16* vTb = vT + (long)b * INTER * SEQ;
  // wave handles contiguous 24 n-tiles as 6 groups of 4
  for (int ng = 0; ng < 6; ++ng) {
    int ntb = wave * 24 + ng * 4;
    v8f c[4];
#pragma unroll
    for (int j4 = 0; j4 < 4; ++j4) c[j4] = (v8f){0.f, 0.f, 0.f, 0.f, 0.f, 0.f, 0.f, 0.f};
    for (int jp = 0; jp < SEQ / 32; ++jp) {
      v16bf a = load_frag_a(pbuf, SEQ, 0, jp * 32, lane);
#pragma unroll
      for (int j4 = 0; j4 < 4; ++j4) {
        v16bf bb = load_frag_b(vTb, SEQ, (ntb + j4) * 16, jp * 32, lane);
        c[j4] = WMMA_BF16(a, bb, c[j4]);
      }
    }
#pragma unroll
    for (int j4 = 0; j4 < 4; ++j4) {
      int col = (ntb + j4) * 16 + ln;
#pragma unroll
      for (int r = 0; r < 8; ++r) {
        long row = m0 + r + 8 * h;
        float uu = u[row * (long)INTER + col];
        uc[row * (long)INTER + col] = f2bf(c[j4][r] * uu);
      }
    }
  }
}

// ---------------- GEMM2: out = (u*ctx) @ Wo + bo ----------------
// One wave computes a 32x64 tile (2 M-tiles x 4 N-tiles).

__global__ __launch_bounds__(32) void k_gemm2(
    const BF16* __restrict__ uc, const BF16* __restrict__ WoT,
    const float* __restrict__ bo, float* __restrict__ out) {
  int lane = threadIdx.x;
  int n0 = blockIdx.x * 64;
  long m0 = (long)blockIdx.y * 32;
  int h = lane >> 4, ln = lane & 15;

  v8f acc[2][4];
#pragma unroll
  for (int i = 0; i < 2; ++i)
#pragma unroll
    for (int j = 0; j < 4; ++j) {
      float bias = bo[n0 + j * 16 + ln];
      acc[i][j] = (v8f){bias, bias, bias, bias, bias, bias, bias, bias};
    }

  for (int k0 = 0; k0 < INTER; k0 += 32) {
    v16bf a0 = load_frag_a(uc, INTER, m0, k0, lane);
    v16bf a1 = load_frag_a(uc, INTER, m0 + 16, k0, lane);
    if (k0 + 64 < INTER) {
      __builtin_prefetch(uc + (m0 + (lane & 15)) * (long)INTER + k0 + 64, 0, 1);
      __builtin_prefetch(WoT + (n0 + (lane & 15)) * (long)INTER + k0 + 64, 0, 1);
    }
#pragma unroll
    for (int j = 0; j < 4; ++j) {
      v16bf b = load_frag_b(WoT, INTER, n0 + j * 16, k0, lane);
      acc[0][j] = WMMA_BF16(a0, b, acc[0][j]);
      acc[1][j] = WMMA_BF16(a1, b, acc[1][j]);
    }
  }

#pragma unroll
  for (int i = 0; i < 2; ++i) {
    long rbase = m0 + i * 16 + 8 * h;
#pragma unroll
    for (int j = 0; j < 4; ++j) {
      int col = n0 + j * 16 + ln;
#pragma unroll
      for (int r = 0; r < 8; ++r)
        out[(rbase + r) * (long)HIDDEN + col] = acc[i][j][r];
    }
  }
}

// ---------------- host launcher ----------------

extern "C" void kernel_launch(void* const* d_in, const int* in_sizes, int n_in,
                              void* d_out, int out_size, void* d_ws, size_t ws_size,
                              hipStream_t stream) {
  (void)in_sizes; (void)n_in; (void)out_size; (void)ws_size;
  const float* hidden = (const float*)d_in[0];
  const int*   maskp  = (const int*)d_in[1];
  /* d_in[2] = position_ids: unused by reference */
  const float* Wi    = (const float*)d_in[3];
  const float* bi    = (const float*)d_in[4];
  const float* gamma = (const float*)d_in[5];
  const float* beta  = (const float*)d_in[6];
  const float* Wo    = (const float*)d_in[7];
  const float* bo    = (const float*)d_in[8];
  float* out = (float*)d_out;

  char* w = (char*)d_ws;
  auto alloc = [&](size_t bytes) {
    char* p = w;
    w += (bytes + 255) & ~(size_t)255;
    return p;
  };
  BF16*  Xb     = (BF16*)alloc((size_t)NTOK * HIDDEN * 2);
  BF16*  WiT    = (BF16*)alloc((size_t)NI * HIDDEN * 2);
  BF16*  WoT    = (BF16*)alloc((size_t)HIDDEN * INTER * 2);
  float* u      = (float*)alloc((size_t)NTOK * INTER * 4);
  BF16*  vT     = (BF16*)alloc((size_t)BATCH * INTER * SEQ * 2);
  BF16*  qb     = (BF16*)alloc((size_t)NTOK * KEY * 2);
  BF16*  kb     = (BF16*)alloc((size_t)NTOK * KEY * 2);
  BF16*  uc     = (BF16*)alloc((size_t)NTOK * INTER * 2);
  float* scaleB = (float*)alloc(BATCH * sizeof(float));

  long nX = (long)NTOK * HIDDEN;
  k_cvt_bf16<<<dim3((unsigned)((nX + 255) / 256)), 256, 0, stream>>>(hidden, Xb, nX);
  long nWi = (long)HIDDEN * NI;
  k_transpose_bf16<<<dim3((unsigned)((nWi + 255) / 256)), 256, 0, stream>>>(Wi, WiT, HIDDEN, NI);
  long nWo = (long)INTER * HIDDEN;
  k_transpose_bf16<<<dim3((unsigned)((nWo + 255) / 256)), 256, 0, stream>>>(Wo, WoT, INTER, HIDDEN);
  k_scale<<<dim3(BATCH), 256, 0, stream>>>(maskp, scaleB);

  k_gemm1<<<dim3(NI / 64, NTOK / 32), 32, 0, stream>>>(Xb, WiT, bi, gamma, beta, u, vT, qb, kb);
  k_attn<<<dim3(SEQ / 16, BATCH), 128, 0, stream>>>(qb, kb, vT, maskp, scaleB, u, uc);
  k_gemm2<<<dim3(HIDDEN / 64, NTOK / 32), 32, 0, stream>>>(uc, WoT, bo, out);
}
